// NNemb_41180146434794
// MI455X (gfx1250) — matrix operand order repaired and checked by reading, not compiled, not run on hardware
//
#include <hip/hip_runtime.h>

// NNConv x2 fused for gfx1250 (MI455X, wave32, WMMA).
// Edge-MLP GEMMs ([E,64]@[64,128] and [E,64]@[64,256]) run as bf16 WMMA with
// fp32 accumulation; per-edge weight tensors are never materialized.
// v3: rolled column-tile loops (B frags rematerialized from LDS, never
//     spilled to scratch), biases staged in LDS, <=~128 VGPRs -> 8 waves/SIMD.

#define NN  50000
#define NE  800000
#define NT  (NE / 16)   // 50000 exact 16-edge tiles

typedef __attribute__((ext_vector_type(16))) __bf16 v16bf;
typedef __attribute__((ext_vector_type(8)))  float  v8f;

union Frag16 { unsigned int u[8]; v16bf v; };

__device__ __forceinline__ unsigned short f2bf(float f) {
    unsigned int u = __float_as_uint(f);
    u += 0x7FFFu + ((u >> 16) & 1u);          // round-to-nearest-even
    return (unsigned short)(u >> 16);
}

// Build one 16x32 bf16 A-fragment in this file's contiguous-K lane layout:
//   lanes 0-15 : row M=lane,    K = {0..15}  (+32*kb)
//   lanes 16-31: row M=lane-16, K = {16..31} (+32*kb)
// dword v packs K = 16*lh + 2*v  and K+1.
// h[e][k] = relu(a_e * w[k] + b[k])  (rank-1 edge-MLP layer, built in regs)
__device__ __forceinline__ v16bf build_a(float a, const float* w, const float* b,
                                         int kb, int lh) {
    Frag16 f;
#pragma unroll
    for (int v = 0; v < 8; ++v) {
        int k = kb * 32 + lh * 16 + 2 * v;
        float h0 = fmaxf(fmaf(a, w[k],     b[k]),     0.0f);
        float h1 = fmaxf(fmaf(a, w[k + 1], b[k + 1]), 0.0f);
        f.u[v] = (unsigned)f2bf(h0) | ((unsigned)f2bf(h1) << 16);
    }
    return f.v;
}

// Load one 32x16 bf16 B-fragment (K rows striped across lane halves,
// dword v packs k = 16*lh + 2v, 2v+1) from the K-transposed LDS image.
__device__ __forceinline__ v16bf load_b(const unsigned int* bp) {
    Frag16 f;
#pragma unroll
    for (int v = 0; v < 8; ++v) f.u[v] = bp[v];
    return f.v;
}

__device__ __forceinline__ v8f splat8(float x) {
    v8f c;
#pragma unroll
    for (int i = 0; i < 8; ++i) c[i] = x;
    return c;
}

// ---------------- kernel 0: prep -------------------------------------------
// hbuf[v][o] = x[v]@root1 + bias1 ; W1b/W2b -> bf16, K-transposed ([col][k])
__global__ void nnemb_prep(const float* __restrict__ x,
                           const float* __restrict__ root1,
                           const float* __restrict__ bias1,
                           const float* __restrict__ W1b,
                           const float* __restrict__ W2b,
                           float* __restrict__ hbuf,
                           unsigned short* __restrict__ w1bT,
                           unsigned short* __restrict__ w2bT) {
    int tid = blockIdx.x * 256 + threadIdx.x;
    if (tid < NN * 64) {
        int v = tid >> 6, o = tid & 63;
        hbuf[tid] = fmaf(x[v * 2 + 0], root1[o],
                    fmaf(x[v * 2 + 1], root1[64 + o], bias1[o]));
    }
    if (tid < 128 * 64) {               // W1b: [64,128] -> bf16 [col][k]
        int col = tid >> 6, k = tid & 63;
        w1bT[col * 64 + k] = f2bf(W1b[k * 128 + col]);
    }
    if (tid < 256 * 64) {               // W2b: [64,256] -> bf16 [col][k]
        int col = tid >> 6, k = tid & 63;
        w2bT[col * 64 + k] = f2bf(W2b[k * 256 + col]);
    }
}

// ---------------- kernel 1: conv1 edge pass --------------------------------
// Per 16-edge tile (one wave): A = relu(a*W1a+b1a) [16x64] bf16 (registers).
// Output columns processed in a ROLLED loop of pairs (tc, tc+4): only two
// 16x16 C-fragments are ever live, B fragments are transient LDS reads.
// msg[:, 16tc..] = x0*U[:, 16tc..] + x1*U[:, 64+16tc..], scattered with
// fp32 atomics into hbuf[dst].
__global__ void __launch_bounds__(256, 4)
nnemb_conv1(const float* __restrict__ x, const int* __restrict__ ei,
            const float* __restrict__ ea, const float* __restrict__ W1a,
            const float* __restrict__ b1a,
            const unsigned short* __restrict__ w1bT,
            const float* __restrict__ b1b, float* __restrict__ hbuf) {
    __shared__ unsigned int sW[4096];   // W1b bf16 [col 0..127][k 0..63], pair-packed
    __shared__ float sWa[64], sBa[64], sBb[128];
    for (int i = threadIdx.x; i < 4096; i += 256)
        sW[i] = ((const unsigned int*)w1bT)[i];
    if (threadIdx.x < 64) { sWa[threadIdx.x] = W1a[threadIdx.x];
                            sBa[threadIdx.x] = b1a[threadIdx.x]; }
    if (threadIdx.x < 128) sBb[threadIdx.x] = b1b[threadIdx.x];
    __syncthreads();

    const int lane = threadIdx.x & 31;
    const int lh   = lane >> 4;         // lane half
    const int nrow = lane & 15;
    const int wave = threadIdx.x >> 5;
    const int nwav = gridDim.x * 8;

    for (int t = blockIdx.x * 8 + wave; t < NT; t += nwav) {
        int   e   = t * 16 + nrow;
        float a   = ea[e];
        int   src = ei[e];
        int   dst = ei[NE + e];
        float x0  = x[src * 2 + 0];
        float x1  = x[src * 2 + 1];

        int tn = t + nwav;              // cover next tile's cold misses
        if (tn < NT) {
            __builtin_prefetch(&ea[tn * 16 + nrow], 0, 3);
            __builtin_prefetch(&ei[tn * 16 + nrow], 0, 3);
            __builtin_prefetch(&ei[NE + tn * 16 + nrow], 0, 3);
        }

        v16bf A0 = build_a(a, sWa, sBa, 0, lh);
        v16bf A1 = build_a(a, sWa, sBa, 1, lh);

        // broadcast per-row edge scalars once (row m lives in lane m)
        float x0m[8], x1m[8];
        int dstm[8];
#pragma unroll
        for (int r = 0; r < 8; ++r) {
            int m   = r + lh * 8;
            x0m[r]  = __shfl(x0, m, 32);
            x1m[r]  = __shfl(x1, m, 32);
            dstm[r] = __shfl(dst, m, 32);
        }

#pragma unroll 1                        // ROLLED: keep live set small, no spills
        for (int tc = 0; tc < 4; ++tc) {
            int colA = tc * 16 + nrow;          // u[:, 0:64]  piece
            int colB = colA + 64;               // u[:, 64:128] piece
            v8f cA = splat8(sBb[colA]);
            v8f cB = splat8(sBb[colB]);
            const unsigned int* bpA = &sW[colA * 32 + lh * 8];
            const unsigned int* bpB = &sW[colB * 32 + lh * 8];
            cA = __builtin_amdgcn_wmma_f32_16x16x32_bf16(false, A0, false,
                     load_b(bpA), (short)0, cA, false, false);
            cB = __builtin_amdgcn_wmma_f32_16x16x32_bf16(false, A0, false,
                     load_b(bpB), (short)0, cB, false, false);
            cA = __builtin_amdgcn_wmma_f32_16x16x32_bf16(false, A1, false,
                     load_b(bpA + 16), (short)0, cA, false, false);
            cB = __builtin_amdgcn_wmma_f32_16x16x32_bf16(false, A1, false,
                     load_b(bpB + 16), (short)0, cB, false, false);
            // D layout: vgpr r -> row M = r + 8*lh, col N = nrow
#pragma unroll
            for (int r = 0; r < 8; ++r) {
                float val = fmaf(x0m[r], cA[r], x1m[r] * cB[r]);
                atomicAdd(&hbuf[dstm[r] * 64 + colA], val);
            }
        }
    }
}

// ---------------- kernel 2: relu + root2 path ------------------------------
__global__ void nnemb_relu_root2(float* __restrict__ hbuf,
                                 const float* __restrict__ root2,
                                 const float* __restrict__ bias2,
                                 float* __restrict__ out) {
    int v = blockIdx.x * 256 + threadIdx.x;
    if (v >= NN) return;
    float a0 = bias2[0], a1 = bias2[1], a2 = bias2[2], a3 = bias2[3];
#pragma unroll 8
    for (int k = 0; k < 64; ++k) {
        float hv = fmaxf(hbuf[v * 64 + k], 0.0f);
        hbuf[v * 64 + k] = hv;                       // relu in place for conv2 gathers
        a0 = fmaf(hv, root2[k * 4 + 0], a0);
        a1 = fmaf(hv, root2[k * 4 + 1], a1);
        a2 = fmaf(hv, root2[k * 4 + 2], a2);
        a3 = fmaf(hv, root2[k * 4 + 3], a3);
    }
    out[v * 4 + 0] = a0; out[v * 4 + 1] = a1;
    out[v * 4 + 2] = a2; out[v * 4 + 3] = a3;
}

// ---------------- kernel 3: conv2 edge pass --------------------------------
// U[16x256] = relu(a*W2a+b2a) @ W2b + b2b, one 16x16 col-tile at a time.
// Col c of U = (k = c>>2, o = c&3); each tile is contracted immediately
// against gathered h[src][k] (L2-resident), acc reduced over the 4 k-groups
// via shfl_xor, then 4 fp32 atomics per edge into out[dst].
__global__ void __launch_bounds__(256, 4)
nnemb_conv2(const float* __restrict__ hbuf, const int* __restrict__ ei,
            const float* __restrict__ ea, const float* __restrict__ W2a,
            const float* __restrict__ b2a,
            const unsigned short* __restrict__ w2bT,
            const float* __restrict__ b2b, float* __restrict__ out) {
    __shared__ unsigned int sW[8192];   // W2b bf16 [col 0..255][k 0..63]
    __shared__ float sWa[64], sBa[64], sBb[256];
    for (int i = threadIdx.x; i < 8192; i += 256)
        sW[i] = ((const unsigned int*)w2bT)[i];
    if (threadIdx.x < 64) { sWa[threadIdx.x] = W2a[threadIdx.x];
                            sBa[threadIdx.x] = b2a[threadIdx.x]; }
    sBb[threadIdx.x] = b2b[threadIdx.x];
    __syncthreads();

    const int lane = threadIdx.x & 31;
    const int lh   = lane >> 4;
    const int nrow = lane & 15;
    const int kk   = nrow >> 2;         // k sub-index inside col-tile
    const int oo   = nrow & 3;          // output channel this lane feeds
    const int wave = threadIdx.x >> 5;
    const int nwav = gridDim.x * 8;

    for (int t = blockIdx.x * 8 + wave; t < NT; t += nwav) {
        int   e   = t * 16 + nrow;
        float a   = ea[e];
        int   src = ei[e];
        int   dst = ei[NE + e];

        int tn = t + nwav;
        if (tn < NT) {
            __builtin_prefetch(&ea[tn * 16 + nrow], 0, 3);
            __builtin_prefetch(&ei[tn * 16 + nrow], 0, 3);
            __builtin_prefetch(&ei[NE + tn * 16 + nrow], 0, 3);
        }

        v16bf A0 = build_a(a, sWa, sBa, 0, lh);
        v16bf A1 = build_a(a, sWa, sBa, 1, lh);

        int srcm[8];
#pragma unroll
        for (int r = 0; r < 8; ++r) srcm[r] = __shfl(src, r + lh * 8, 32);

        float acc[8] = {0, 0, 0, 0, 0, 0, 0, 0};
#pragma unroll 2                        // bounded ILP: ~2 B/C sets live
        for (int n = 0; n < 16; ++n) {
            int col = n * 16 + nrow;
            v8f c = splat8(sBb[col]);
            const unsigned int* bp = &sW[col * 32 + lh * 8];
            c = __builtin_amdgcn_wmma_f32_16x16x32_bf16(false, A0, false,
                    load_b(bp), (short)0, c, false, false);
            c = __builtin_amdgcn_wmma_f32_16x16x32_bf16(false, A1, false,
                    load_b(bp + 16), (short)0, c, false, false);
            int k = n * 4 + kk;
#pragma unroll
            for (int r = 0; r < 8; ++r)
                acc[r] = fmaf(hbuf[srcm[r] * 64 + k], c[r], acc[r]);
        }

#pragma unroll
        for (int r = 0; r < 8; ++r) {
            float v = acc[r];
            v += __shfl_xor(v, 4, 32);               // reduce over the 4 k-groups
            v += __shfl_xor(v, 8, 32);
            int dstm = __shfl(dst, r + lh * 8, 32);
            if (nrow < 4) atomicAdd(&out[dstm * 4 + oo], v);
        }
    }
}

extern "C" void kernel_launch(void* const* d_in, const int* in_sizes, int n_in,
                              void* d_out, int out_size, void* d_ws, size_t ws_size,
                              hipStream_t stream) {
    const float* x     = (const float*)d_in[0];
    const int*   ei    = (const int*)  d_in[1];
    const float* ea    = (const float*)d_in[2];
    const float* W1a   = (const float*)d_in[3];
    const float* b1a   = (const float*)d_in[4];
    const float* W1b   = (const float*)d_in[5];
    const float* b1b   = (const float*)d_in[6];
    const float* root1 = (const float*)d_in[7];
    const float* bias1 = (const float*)d_in[8];
    const float* W2a   = (const float*)d_in[9];
    const float* b2a   = (const float*)d_in[10];
    const float* W2b   = (const float*)d_in[11];
    const float* b2b   = (const float*)d_in[12];
    const float* root2 = (const float*)d_in[13];
    const float* bias2 = (const float*)d_in[14];

    float* out  = (float*)d_out;
    float* hbuf = (float*)d_ws;                                   // 50000*64 f32
    unsigned short* w1bT = (unsigned short*)((char*)d_ws + (size_t)NN * 64 * 4);
    unsigned short* w2bT = w1bT + 128 * 64;

    nnemb_prep<<<(NN * 64 + 255) / 256, 256, 0, stream>>>(
        x, root1, bias1, W1b, W2b, hbuf, w1bT, w2bT);
    nnemb_conv1<<<1024, 256, 0, stream>>>(
        x, ei, ea, W1a, b1a, w1bT, b1b, hbuf);
    nnemb_relu_root2<<<(NN + 255) / 256, 256, 0, stream>>>(
        hbuf, root2, bias2, out);
    nnemb_conv2<<<1024, 256, 0, stream>>>(
        hbuf, ei, ea, W2a, b2a, w2bT, b2b, out);
}
